// Attention_88416196755838
// MI455X (gfx1250) — matrix operand order repaired and checked by reading, not compiled, not run on hardware
//
#include <hip/hip_runtime.h>
#include <hip/hip_bf16.h>

// Problem constants (B=1)
constexpr int S  = 2048;
constexpr int D  = 4096;
constexpr int H  = 32;
constexpr int HD = 128;

typedef __attribute__((ext_vector_type(16))) __bf16 v16bf;
typedef __attribute__((ext_vector_type(8)))  float  v8f;

union Frag {
    v16bf v;
    unsigned int u[8];
};

__device__ __forceinline__ unsigned short f2bf(float f) {
    unsigned int u = __float_as_uint(f);
    unsigned int r = (u + 0x7FFFu + ((u >> 16) & 1u)) >> 16;  // RNE
    return (unsigned short)r;
}

__device__ __forceinline__ unsigned int pack2(float a, float b) {
    return (unsigned int)f2bf(a) | ((unsigned int)f2bf(b) << 16);
}

// gfx1250 async global->LDS copy (ASYNCcnt-tracked); per-lane LDS dest offset
// in a VGPR, per-lane 64-bit global address in a VGPR pair.
__device__ __forceinline__ void async_copy_b128(unsigned lds_off, const void* gaddr) {
    asm volatile("global_load_async_to_lds_b128 %0, %1, off"
                 :: "v"(lds_off), "v"(gaddr)
                 : "memory");
}
__device__ __forceinline__ void wait_async0() {
    asm volatile("s_wait_asynccnt 0x0" ::: "memory");
}

// ---------------------------------------------------------------------------
// GEMM: C[M=S, N=D] = A[S, D] @ W[D, D], bf16 WMMA, f32 accumulation.
// 128x128 block tile, 8 waves (4M x 2N), each wave 32x64 = 2x4 WMMA tiles.
// Double-buffered LDS; W staged transposed with packed u32 stores.
// (f32->bf16 conversion happens during staging, so this path keeps the
//  VGPR round trip by necessity.)
// ---------------------------------------------------------------------------
template <bool BF16OUT>
__global__ __launch_bounds__(256)
void gemm_kernel(const float* __restrict__ A, const float* __restrict__ W,
                 float* __restrict__ outF, unsigned short* __restrict__ outB)
{
    constexpr int K = D, N = D;
    constexpr int NK = K / 32;
    __shared__ unsigned short As[2][128 * 36];  // [row][k], stride 36
    __shared__ unsigned short Bs[2][128 * 36];  // transposed [n][k], stride 36

    const int tid  = threadIdx.x;
    const int wave = tid >> 5, lane = tid & 31;
    const int half = lane >> 4, lm = lane & 15;
    const int waveM = wave & 3, waveN = wave >> 2;
    const int m0 = blockIdx.x * 128, n0 = blockIdx.y * 128;

    int arow[4], ac4[4];
#pragma unroll
    for (int i = 0; i < 4; ++i) {
        int e = tid + i * 256;                // A tile: 128 rows x 32 cols
        arow[i] = e >> 3;
        ac4[i]  = (e & 7) * 4;
    }
    int wkp[2], wcg[2];
#pragma unroll
    for (int i = 0; i < 2; ++i) {
        int e = tid + i * 256;                // W tile: 16 row-pairs x 32 col-groups
        wkp[i] = e >> 5;
        wcg[i] = (e & 31) * 4;
    }

    float4 aReg[4], wReg[2][2];

    auto load_tiles = [&](int k0) {
#pragma unroll
        for (int i = 0; i < 4; ++i)
            aReg[i] = *(const float4*)&A[(size_t)(m0 + arow[i]) * K + k0 + ac4[i]];
#pragma unroll
        for (int i = 0; i < 2; ++i) {
            wReg[i][0] = *(const float4*)&W[(size_t)(k0 + 2 * wkp[i] + 0) * N + n0 + wcg[i]];
            wReg[i][1] = *(const float4*)&W[(size_t)(k0 + 2 * wkp[i] + 1) * N + n0 + wcg[i]];
        }
    };
    auto store_tiles = [&](int buf) {
#pragma unroll
        for (int i = 0; i < 4; ++i) {
            unsigned int* dst = (unsigned int*)&As[buf][arow[i] * 36 + ac4[i]];
            dst[0] = pack2(aReg[i].x, aReg[i].y);
            dst[1] = pack2(aReg[i].z, aReg[i].w);
        }
#pragma unroll
        for (int i = 0; i < 2; ++i) {
            float a0[4] = {wReg[i][0].x, wReg[i][0].y, wReg[i][0].z, wReg[i][0].w};
            float a1[4] = {wReg[i][1].x, wReg[i][1].y, wReg[i][1].z, wReg[i][1].w};
#pragma unroll
            for (int j = 0; j < 4; ++j)
                *(unsigned int*)&Bs[buf][(wcg[i] + j) * 36 + 2 * wkp[i]] = pack2(a0[j], a1[j]);
        }
    };

    v8f acc[2][4];
#pragma unroll
    for (int mt = 0; mt < 2; ++mt)
#pragma unroll
        for (int nt = 0; nt < 4; ++nt)
            acc[mt][nt] = (v8f){0.f, 0.f, 0.f, 0.f, 0.f, 0.f, 0.f, 0.f};

    load_tiles(0);
    store_tiles(0);

    for (int kt = 0; kt < NK; ++kt) {
        __syncthreads();
        const int cur = kt & 1;
        if (kt + 1 < NK) {
            load_tiles((kt + 1) * 32);
            if (kt + 2 < NK) {
                __builtin_prefetch(&A[(size_t)(m0 + arow[0]) * K + (kt + 2) * 32 + ac4[0]], 0, 3);
                __builtin_prefetch(&W[(size_t)((kt + 2) * 32 + 2 * wkp[0]) * N + n0 + wcg[0]], 0, 3);
            }
        }

        const unsigned short* as = As[cur];
        const unsigned short* bs = Bs[cur];

        Frag af[2], bfr[4];
#pragma unroll
        for (int mt = 0; mt < 2; ++mt) {
            int row = waveM * 32 + mt * 16 + lm;
#pragma unroll
            for (int v = 0; v < 8; ++v) {
                int k = ((v & 4) << 2) + half * 8 + 2 * (v & 3);
                af[mt].u[v] = *(const unsigned int*)&as[row * 36 + k];
            }
        }
#pragma unroll
        for (int nt = 0; nt < 4; ++nt) {
            int col = waveN * 64 + nt * 16 + lm;
#pragma unroll
            for (int v = 0; v < 8; ++v) {
                int k = ((v & 4) << 2) + half * 8 + 2 * (v & 3);
                bfr[nt].u[v] = *(const unsigned int*)&bs[col * 36 + k];
            }
        }
#pragma unroll
        for (int nt = 0; nt < 4; ++nt)
#pragma unroll
            for (int mt = 0; mt < 2; ++mt)
                acc[mt][nt] = __builtin_amdgcn_wmma_f32_16x16x32_bf16(
                    false, af[mt].v, false, bfr[nt].v, (short)0, acc[mt][nt], false, false);

        if (kt + 1 < NK) store_tiles(cur ^ 1);
    }

#pragma unroll
    for (int mt = 0; mt < 2; ++mt)
#pragma unroll
        for (int nt = 0; nt < 4; ++nt)
#pragma unroll
            for (int r = 0; r < 8; ++r) {
                int row = m0 + waveM * 32 + mt * 16 + half * 8 + r;
                int col = n0 + waveN * 64 + nt * 16 + lm;
                if (BF16OUT)
                    outB[(size_t)row * N + col] = f2bf(acc[mt][nt][r]);
                else
                    outF[(size_t)row * N + col] = acc[mt][nt][r];
            }
}

// ---------------------------------------------------------------------------
// RoPE over full model dim (reference applies it before the head split).
// One thread handles the coupled pair (d, d+D/2); writes bf16.
// ---------------------------------------------------------------------------
__global__ __launch_bounds__(256)
void rope_kernel(const float* __restrict__ X, const float* __restrict__ cosT,
                 const float* __restrict__ sinT, unsigned short* __restrict__ out)
{
    int i = blockIdx.x * 256 + threadIdx.x;  // over S * (D/2)
    int s = i >> 11;                         // / (D/2)
    int d = i & (D / 2 - 1);
    size_t i1 = (size_t)s * D + d;
    size_t i2 = i1 + D / 2;
    float x1 = X[i1], x2 = X[i2];
    out[i1] = f2bf(x1 * cosT[i1] - x2 * sinT[i1]);
    out[i2] = f2bf(x2 * cosT[i2] + x1 * sinT[i2]);
}

// ---------------------------------------------------------------------------
// Flash attention: grid = (S/128 query tiles, H). 8 waves x 16 query rows.
// Double-buffered K/V tiles. K tile staged via async global->LDS B128 copies
// (ASYNCcnt); V tile staged transposed with packed u32 LDS stores.
// ---------------------------------------------------------------------------
__global__ __launch_bounds__(256)
void attn_kernel(const unsigned short* __restrict__ Qb,
                 const unsigned short* __restrict__ Kb,
                 const unsigned short* __restrict__ Vb,
                 const float* __restrict__ amask,
                 float* __restrict__ O)
{
    constexpr int NT = S / 64;                 // 32 key tiles
    __shared__ unsigned short Kt[2][64 * 136]; // [key][hd], stride 136
    __shared__ unsigned short Vt[2][128 * 68]; // transposed [hd][key], stride 68
    __shared__ unsigned short Pw[8 * 16 * 68]; // per-wave P staging [16][64+pad]

    const int tid  = threadIdx.x;
    const int wave = tid >> 5, lane = tid & 31;
    const int half = lane >> 4, lm = lane & 15;
    const int qrow0 = blockIdx.x * 128 + wave * 16;
    const int hoff  = blockIdx.y * HD;
    const float scale = 1.0f / 64.0f;          // 1/sqrt(D), per reference

    int kkey[4], kc8[4];
#pragma unroll
    for (int i = 0; i < 4; ++i) {
        int e = tid + i * 256;                 // K tile: 64 keys x 16 groups of 8
        kkey[i] = e >> 4;
        kc8[i]  = (e & 15) * 8;
    }
    int vkp[2], vc8[2];
#pragma unroll
    for (int i = 0; i < 2; ++i) {
        int e = tid + i * 256;                 // V tile: 32 key-pairs x 16 groups of 8
        vkp[i] = e >> 4;
        vc8[i] = (e & 15) * 8;
    }

    // Async K-tile staging: pure strided copy, no VGPR round trip.
    auto async_k = [&](int buf, int k0) {
#pragma unroll
        for (int i = 0; i < 4; ++i)
            async_copy_b128((unsigned)(size_t)&Kt[buf][kkey[i] * 136 + kc8[i]],
                            &Kb[(size_t)(k0 + kkey[i]) * D + hoff + kc8[i]]);
    };

    uint4 vReg[2][2];
    auto load_v = [&](int k0) {
#pragma unroll
        for (int i = 0; i < 2; ++i) {
            vReg[i][0] = *(const uint4*)&Vb[(size_t)(k0 + 2 * vkp[i] + 0) * D + hoff + vc8[i]];
            vReg[i][1] = *(const uint4*)&Vb[(size_t)(k0 + 2 * vkp[i] + 1) * D + hoff + vc8[i]];
        }
    };
    auto store_v = [&](int buf) {
#pragma unroll
        for (int i = 0; i < 2; ++i) {
            const unsigned short* p0 = (const unsigned short*)&vReg[i][0];
            const unsigned short* p1 = (const unsigned short*)&vReg[i][1];
#pragma unroll
            for (int j = 0; j < 8; ++j)
                *(unsigned int*)&Vt[buf][(vc8[i] + j) * 68 + 2 * vkp[i]] =
                    (unsigned int)p0[j] | ((unsigned int)p1[j] << 16);
        }
    };

    // Q fragments: 16 rows x 128 (4 chunks of K=32), in registers
    Frag qf[4];
#pragma unroll
    for (int c = 0; c < 4; ++c)
#pragma unroll
        for (int v = 0; v < 8; ++v) {
            int k = c * 32 + ((v & 4) << 2) + half * 8 + 2 * (v & 3);
            qf[c].u[v] = *(const unsigned int*)&Qb[(size_t)(qrow0 + lm) * D + hoff + k];
        }

    v8f oacc[8];
#pragma unroll
    for (int nt = 0; nt < 8; ++nt)
        oacc[nt] = (v8f){0.f, 0.f, 0.f, 0.f, 0.f, 0.f, 0.f, 0.f};
    float mstate[8], lstate[8];
#pragma unroll
    for (int r = 0; r < 8; ++r) { mstate[r] = -3.0e38f; lstate[r] = 0.f; }

    unsigned short* pw = &Pw[wave * 16 * 68];

    async_k(0, 0);
    load_v(0);
    store_v(0);

    for (int it = 0; it < NT; ++it) {
        wait_async0();          // K tile for this iteration landed
        __syncthreads();        // ...and is published block-wide
        const int cur = it & 1;
        const int k0  = it * 64;
        if (it + 1 < NT) {
            async_k(cur ^ 1, (it + 1) * 64);
            load_v((it + 1) * 64);
            if (it + 2 < NT)
                __builtin_prefetch(&Vb[(size_t)((it + 2) * 64 + 2 * vkp[0]) * D + hoff + vc8[0]], 0, 3);
        }
        const unsigned short* kt = Kt[cur];
        const unsigned short* vt = Vt[cur];

        // S = Q @ K^T: chunk-outer so the 4 WMMAs per chunk are independent
        v8f sacc[4];
#pragma unroll
        for (int nt = 0; nt < 4; ++nt)
            sacc[nt] = (v8f){0.f, 0.f, 0.f, 0.f, 0.f, 0.f, 0.f, 0.f};
#pragma unroll
        for (int c = 0; c < 4; ++c) {
            Frag bfr[4];
#pragma unroll
            for (int nt = 0; nt < 4; ++nt)
#pragma unroll
                for (int v = 0; v < 8; ++v) {
                    int kd = c * 32 + ((v & 4) << 2) + half * 8 + 2 * (v & 3);
                    bfr[nt].u[v] = *(const unsigned int*)&kt[(nt * 16 + lm) * 136 + kd];
                }
#pragma unroll
            for (int nt = 0; nt < 4; ++nt)
                sacc[nt] = __builtin_amdgcn_wmma_f32_16x16x32_bf16(
                    false, qf[c].v, false, bfr[nt].v, (short)0, sacc[nt], false, false);
        }

        // Scale + mask IN PLACE (sacc is dead as an accumulator now), row max
        float mt_[8];
#pragma unroll
        for (int r = 0; r < 8; ++r) mt_[r] = -3.0e38f;
#pragma unroll
        for (int nt = 0; nt < 4; ++nt)
#pragma unroll
            for (int r = 0; r < 8; ++r) {
                int kcol = k0 + nt * 16 + lm;
                int qrow = qrow0 + half * 8 + r;
                float mv = amask[(size_t)qrow * S + kcol] +
                           ((kcol <= qrow) ? -1.0e9f : 0.0f);
                float s = sacc[nt][r] * scale + mv;
                sacc[nt][r] = s;
                mt_[r] = fmaxf(mt_[r], s);
            }
#pragma unroll
        for (int off = 1; off <= 8; off <<= 1)
#pragma unroll
            for (int r = 0; r < 8; ++r)
                mt_[r] = fmaxf(mt_[r], __shfl_xor(mt_[r], off, 32));

        float alpha[8];
#pragma unroll
        for (int r = 0; r < 8; ++r) {
            float mnew = fmaxf(mstate[r], mt_[r]);
            alpha[r] = __expf(mstate[r] - mnew);
            mstate[r] = mnew;
        }

        float ls[8];
#pragma unroll
        for (int r = 0; r < 8; ++r) ls[r] = 0.f;
#pragma unroll
        for (int nt = 0; nt < 4; ++nt)
#pragma unroll
            for (int r = 0; r < 8; ++r) {
                float p = __expf(sacc[nt][r] - mstate[r]);
                ls[r] += p;
                pw[(half * 8 + r) * 68 + nt * 16 + lm] = f2bf(p);
            }
#pragma unroll
        for (int off = 1; off <= 8; off <<= 1)
#pragma unroll
            for (int r = 0; r < 8; ++r)
                ls[r] += __shfl_xor(ls[r], off, 32);
#pragma unroll
        for (int r = 0; r < 8; ++r)
            lstate[r] = lstate[r] * alpha[r] + ls[r];
#pragma unroll
        for (int nt = 0; nt < 8; ++nt)
#pragma unroll
            for (int r = 0; r < 8; ++r)
                oacc[nt][r] *= alpha[r];

        // O += P @ V: group 4 B-fragment loads, then 4 independent WMMAs
#pragma unroll
        for (int c2 = 0; c2 < 2; ++c2) {
            Frag pf;
#pragma unroll
            for (int v = 0; v < 8; ++v) {
                int kk = c2 * 32 + ((v & 4) << 2) + half * 8 + 2 * (v & 3);
                pf.u[v] = *(const unsigned int*)&pw[lm * 68 + kk];
            }
#pragma unroll
            for (int ng = 0; ng < 2; ++ng) {
                Frag vf[4];
#pragma unroll
                for (int t = 0; t < 4; ++t)
#pragma unroll
                    for (int v = 0; v < 8; ++v) {
                        int kk = c2 * 32 + ((v & 4) << 2) + half * 8 + 2 * (v & 3);
                        vf[t].u[v] =
                            *(const unsigned int*)&vt[((ng * 4 + t) * 16 + lm) * 68 + kk];
                    }
#pragma unroll
                for (int t = 0; t < 4; ++t)
                    oacc[ng * 4 + t] = __builtin_amdgcn_wmma_f32_16x16x32_bf16(
                        false, pf.v, false, vf[t].v, (short)0, oacc[ng * 4 + t], false, false);
            }
        }

        if (it + 1 < NT) store_v(cur ^ 1);
    }

    // Normalize and write per-head output into O[s][h*128 + col]
#pragma unroll
    for (int nt = 0; nt < 8; ++nt)
#pragma unroll
        for (int r = 0; r < 8; ++r) {
            int row = qrow0 + half * 8 + r;
            int col = hoff + nt * 16 + lm;
            O[(size_t)row * D + col] = oacc[nt][r] / lstate[r];
        }
}

// ---------------------------------------------------------------------------
// Host-side launch
// ---------------------------------------------------------------------------
extern "C" void kernel_launch(void* const* d_in, const int* in_sizes, int n_in,
                              void* d_out, int out_size, void* d_ws, size_t ws_size,
                              hipStream_t stream)
{
    const float* hidden = (const float*)d_in[0];
    const float* amask  = (const float*)d_in[1];
    const float* cosT   = (const float*)d_in[2];
    const float* sinT   = (const float*)d_in[3];
    const float* wq     = (const float*)d_in[4];
    const float* wk     = (const float*)d_in[5];
    const float* wv     = (const float*)d_in[6];
    const float* wo     = (const float*)d_in[7];
    float* out = (float*)d_out;

    char* ws = (char*)d_ws;
    const size_t fsz = (size_t)S * D * sizeof(float);          // 32 MB
    const size_t hsz = (size_t)S * D * sizeof(unsigned short); // 16 MB
    float*          Qf   = (float*)(ws);
    float*          Kf   = (float*)(ws + fsz);
    unsigned short* Qb   = (unsigned short*)(ws + 2 * fsz);
    unsigned short* Kb   = (unsigned short*)(ws + 2 * fsz + hsz);
    unsigned short* Vb   = (unsigned short*)(ws + 2 * fsz + 2 * hsz);
    float*          Oacc = (float*)(ws + 2 * fsz + 3 * hsz);

    dim3 gg(S / 128, D / 128);   // 16 x 32
    dim3 bb(256);

    gemm_kernel<false><<<gg, bb, 0, stream>>>(hidden, wq, Qf, nullptr);
    gemm_kernel<false><<<gg, bb, 0, stream>>>(hidden, wk, Kf, nullptr);
    gemm_kernel<true ><<<gg, bb, 0, stream>>>(hidden, wv, nullptr, Vb);

    rope_kernel<<<(S * (D / 2)) / 256, 256, 0, stream>>>(Qf, cosT, sinT, Qb);
    rope_kernel<<<(S * (D / 2)) / 256, 256, 0, stream>>>(Kf, cosT, sinT, Kb);

    attn_kernel<<<dim3(S / 128, H), bb, 0, stream>>>(Qb, Kb, Vb, amask, Oacc);

    gemm_kernel<false><<<gg, bb, 0, stream>>>(Oacc, wo, out, nullptr);
}
